// DirectPanoramaVoxGO_7241314861251
// MI455X (gfx1250) — matrix operand order repaired and checked by reading, not compiled, not run on hardware
//
#include <hip/hip_runtime.h>
#include <math.h>

// ---------------- problem constants ----------------
#define NRAYS     4096
#define NSAMP     192
#define NPTS      (NRAYS * NSAMP)          // 786432
#define GRID_N    160
#define EQH       768
#define EQW       1536
#define RGBD      12
#define MLPW      128
#define XYZ_F     10
#define DIR_F     4
#define IN_DEF    90                       // 3+60 + 3+24
#define KPAD0     96                       // IN_DEF padded to 3*32
#define NEAR_T    0.05f
#define FAR_T     3.0f
#define INTERVAL  0.5f
#define ACT_SHIFT (-9.21024037f)           // log(1/(1-1e-4) - 1)
#define FTHRES    1e-4f

// ---------------- WMMA types ----------------
typedef __attribute__((ext_vector_type(16))) _Float16 v16h;
typedef __attribute__((ext_vector_type(8)))  float    v8f;

union Frag16 { v16h h; unsigned u[8]; };

__device__ inline v8f zero8() {
  v8f z = {0.f,0.f,0.f,0.f,0.f,0.f,0.f,0.f};
  return z;
}

// A fragment, 16-bit 16x32 (ISA 7.12.2): lanes 0-15 hold M=lane, K groups
// {0..7,16..23}; lanes 16-31 hold {8..15,24..31}. K pairs are packed, so load
// as 32-bit words from LDS. rowStrideU = row stride in uints (halves/2).
__device__ inline Frag16 load_a(const unsigned* As32, int rowStrideU, int kc, int lane) {
  int m = lane & 15, hi = lane >> 4;
  Frag16 f;
  int base = m * rowStrideU + kc * 16 + hi * 4;
#pragma unroll
  for (int v = 0; v < 4; ++v) f.u[v]     = As32[base + v];       // K = kc*32 + hi*8 + 2v
#pragma unroll
  for (int v = 0; v < 4; ++v) f.u[4 + v] = As32[base + 8 + v];   // K = +16
  return f;
}

// B fragment, 16-bit 32x16: lane = column N (lanes 0-15 K={0..7,16..23}, 16-31 K={8..15,24..31}),
// weights staged transposed [N][Kpad] in LDS so K pairs are contiguous.
__device__ inline Frag16 load_b(const unsigned* Wt32, int kPadU, int nBase, int kc, int lane) {
  int n = nBase + (lane & 15);
  int hi = lane >> 4;
  Frag16 f;
  int base = n * kPadU + kc * 16 + hi * 4;
#pragma unroll
  for (int v = 0; v < 4; ++v) f.u[v]     = Wt32[base + v];       // K = kc*32 + hi*8 + 2v
#pragma unroll
  for (int v = 0; v < 4; ++v) f.u[4 + v] = Wt32[base + 8 + v];   // K = +16
  return f;
}

// Stage f32 weights [K][N] (row-major) -> f16 LDS transposed [Npad][Kpad], zero padded.
__device__ inline void stage_wt(_Float16* dst, const float* __restrict__ src,
                                int K, int N, int Kpad, int Npad, int tid, int nthr) {
  int total = Kpad * Npad;
  for (int i = tid; i < total; i += nthr) {
    int k = i / Npad, n = i - k * Npad;          // n fastest -> coalesced global reads
    float v = (k < K && n < N) ? src[k * N + n] : 0.f;
    dst[n * Kpad + k] = (_Float16)v;
  }
}

// C-tile (f32 16x16) -> bias + ReLU -> f16 activations [16][128] in LDS.
__device__ inline void store_h_relu(_Float16* As, const v8f* acc, const float* __restrict__ bias, int lane) {
  int n0 = lane & 15, hi = lane >> 4;
#pragma unroll
  for (int nt = 0; nt < 8; ++nt) {
    int n = nt * 16 + n0;
    float bv = bias[n];
#pragma unroll
    for (int j = 0; j < 8; ++j) {
      float v = acc[nt][j] + bv;
      As[(j + 8 * hi) * MLPW + n] = (_Float16)(v > 0.f ? v : 0.f);
    }
  }
}

#define WMMA(A, B, C) __builtin_amdgcn_wmma_f32_16x16x32_f16(false, (A), false, (B), (short)0, (C), false, false)

// ================= kernel 1: ray sampling + contraction + density ==============
__global__ __launch_bounds__(256) void sample_density_kernel(
    const float* __restrict__ rays_o, const float* __restrict__ rays_d,
    const float* __restrict__ dgrid, float* __restrict__ contracted,
    float* __restrict__ alpha_out) {
  int p = blockIdx.x * 256 + threadIdx.x;            // grid sized exactly NPTS
  int ray = p / NSAMP, s = p - ray * NSAMP;

  float ox = rays_o[ray * 3 + 0], oy = rays_o[ray * 3 + 1], oz = rays_o[ray * 3 + 2];
  float dx = rays_d[ray * 3 + 0], dy = rays_d[ray * 3 + 1], dz = rays_d[ray * 3 + 2];
  float dn = sqrtf(dx * dx + dy * dy + dz * dz) + 1e-9f;
  dx /= dn; dy /= dn; dz /= dn;

  float t = NEAR_T + (FAR_T - NEAR_T) * ((float)s / (float)(NSAMP - 1));
  float px = ox + t * dx, py = oy + t * dy, pz = oz + t * dz;
  float r = sqrtf(px * px + py * py + pz * pz);
  float sc = (r <= 1.f) ? 1.f : (2.f - 1.f / r) / r;
  float cx = px * sc * 0.5f, cy = py * sc * 0.5f, cz = pz * sc * 0.5f;
  contracted[p * 3 + 0] = cx;
  contracted[p * 3 + 1] = cy;
  contracted[p * 3 + 2] = cz;

  // trilinear sample, align_corners=True; x->W, y->H, z->D (all GRID_N)
  const float scl = 0.5f * (float)(GRID_N - 1);
  float fx = fminf(fmaxf((cx + 1.f) * scl, 0.f), (float)(GRID_N - 1));
  float fy = fminf(fmaxf((cy + 1.f) * scl, 0.f), (float)(GRID_N - 1));
  float fz = fminf(fmaxf((cz + 1.f) * scl, 0.f), (float)(GRID_N - 1));
  int x0 = (int)fx, y0 = (int)fy, z0 = (int)fz;
  int x1 = min(x0 + 1, GRID_N - 1), y1 = min(y0 + 1, GRID_N - 1), z1 = min(z0 + 1, GRID_N - 1);
  float wx = fx - x0, wy = fy - y0, wz = fz - z0;

  #define G(z, y, x) dgrid[((z) * GRID_N + (y)) * GRID_N + (x)]
  float c00 = G(z0, y0, x0) * (1.f - wx) + G(z0, y0, x1) * wx;
  float c01 = G(z0, y1, x0) * (1.f - wx) + G(z0, y1, x1) * wx;
  float c10 = G(z1, y0, x0) * (1.f - wx) + G(z1, y0, x1) * wx;
  float c11 = G(z1, y1, x0) * (1.f - wx) + G(z1, y1, x1) * wx;
  #undef G
  float dens = (c00 * (1.f - wy) + c01 * wy) * (1.f - wz) + (c10 * (1.f - wy) + c11 * wy) * wz;

  float x = dens + ACT_SHIFT;
  float sp = (x > 20.f) ? x : log1pf(__expf(x));
  float a = 1.f - __expf(-sp * INTERVAL);
  alpha_out[p] = (a > FTHRES) ? a : 0.f;
}

// ================= kernel 2: per-ray compositing weights (in place) ============
__global__ __launch_bounds__(256) void composite_weights_kernel(float* __restrict__ wbuf) {
  int ray = blockIdx.x * 256 + threadIdx.x;
  if (ray >= NRAYS) return;
  float T = 1.f;
  for (int s = 0; s < NSAMP; ++s) {
    int p = ray * NSAMP + s;
    float a = wbuf[p];
    float w = a * T;
    wbuf[p] = (w > FTHRES) ? w : 0.f;
    T *= (1.f - a + 1e-10f);
  }
}

// ================= kernel 3: fused WMMA MLPs + equirect sampling ===============
#define TILES_PER_BLOCK 4
// LDS half-element offsets
#define OFF_W0 0                         // [128][96]
#define OFF_W1 (OFF_W0 + 128 * 96)       // [128][128]
#define OFF_W2 (OFF_W1 + 128 * 128)      // [16][128]
#define OFF_C0 (OFF_W2 + 16 * 128)       // [128][32]
#define OFF_C1 (OFF_C0 + 128 * 32)       // [128][128]
#define OFF_C2 (OFF_C1 + 128 * 128)      // [16][128]
#define OFF_AS (OFF_C2 + 16 * 128)       // 8 waves x [16][128]
#define SMEM_HALVES (OFF_AS + 8 * 16 * 128)
#define SMEM_BYTES (SMEM_HALVES * 2)

__global__ __launch_bounds__(256) void mlp_fused_kernel(
    const float* __restrict__ rays_d, const float* __restrict__ equ,
    const float* __restrict__ dW0, const float* __restrict__ db0,
    const float* __restrict__ dW1, const float* __restrict__ db1,
    const float* __restrict__ dW2, const float* __restrict__ db2,
    const float* __restrict__ cW0, const float* __restrict__ cb0,
    const float* __restrict__ cW1, const float* __restrict__ cb1,
    const float* __restrict__ cW2, const float* __restrict__ cb2,
    const float* __restrict__ contracted, float* __restrict__ rgb_out) {
  extern __shared__ __align__(16) char smem_raw[];
  _Float16* smem = (_Float16*)smem_raw;
  _Float16* W0t = smem + OFF_W0;
  _Float16* W1t = smem + OFF_W1;
  _Float16* W2t = smem + OFF_W2;
  _Float16* C0t = smem + OFF_C0;
  _Float16* C1t = smem + OFF_C1;
  _Float16* C2t = smem + OFF_C2;

  int tid = threadIdx.x;
  int wave = tid >> 5;
  int lane = tid & 31;
  int m16 = lane & 15;
  int hi = lane >> 4;
  _Float16* As = smem + OFF_AS + wave * 16 * MLPW;
  unsigned* As32 = (unsigned*)As;
  const unsigned* W0t32 = (const unsigned*)W0t;
  const unsigned* W1t32 = (const unsigned*)W1t;
  const unsigned* W2t32 = (const unsigned*)W2t;
  const unsigned* C0t32 = (const unsigned*)C0t;
  const unsigned* C1t32 = (const unsigned*)C1t;
  const unsigned* C2t32 = (const unsigned*)C2t;

  // stage all weights once per block, f16 transposed [N][Kpad]
  stage_wt(W0t, dW0, IN_DEF, MLPW, KPAD0, MLPW, tid, 256);
  stage_wt(W1t, dW1, MLPW, MLPW, MLPW, MLPW, tid, 256);
  stage_wt(W2t, dW2, MLPW, 2, MLPW, 16, tid, 256);
  stage_wt(C0t, cW0, RGBD, MLPW, 32, MLPW, tid, 256);
  stage_wt(C1t, cW1, MLPW, MLPW, MLPW, MLPW, tid, 256);
  stage_wt(C2t, cW2, MLPW, 3, MLPW, 16, tid, 256);
  __syncthreads();

  for (int it = 0; it < TILES_PER_BLOCK; ++it) {
    int tile = (blockIdx.x * TILES_PER_BLOCK + it) * 8 + wave;
    int p0 = tile * 16;

    // ---- phase 1: positional-encoded features -> As[16][96] (f16) ----
    // Full-wave split: lane handles point (lane&15); hi=0 does xyz-PE freqs 0..6
    // (21 sincos), hi=1 does xyz-PE freqs 7..9 + dir-PE (21 sincos) + pads.
    int p = p0 + m16;
    int ray = p / NSAMP;
    float cx = contracted[p * 3 + 0], cy = contracted[p * 3 + 1], cz = contracted[p * 3 + 2];
    float ddx = rays_d[ray * 3 + 0], ddy = rays_d[ray * 3 + 1], ddz = rays_d[ray * 3 + 2];
    float dn = sqrtf(ddx * ddx + ddy * ddy + ddz * ddz) + 1e-9f;
    ddx /= dn; ddy /= dn; ddz /= dn;

    _Float16* F = As + m16 * KPAD0;
    if (hi == 0) {
      F[0] = (_Float16)cx; F[1] = (_Float16)cy; F[2] = (_Float16)cz;
      float fr = 1.f;
#pragma unroll
      for (int f = 0; f < 7; ++f) {
        float s0, c0, s1, c1, s2, c2;
        __sincosf(cx * fr, &s0, &c0);
        __sincosf(cy * fr, &s1, &c1);
        __sincosf(cz * fr, &s2, &c2);
        F[3 + f * 3 + 0] = (_Float16)s0;  F[33 + f * 3 + 0] = (_Float16)c0;
        F[3 + f * 3 + 1] = (_Float16)s1;  F[33 + f * 3 + 1] = (_Float16)c1;
        F[3 + f * 3 + 2] = (_Float16)s2;  F[33 + f * 3 + 2] = (_Float16)c2;
        fr *= 2.f;
      }
    } else {
      float fr = 128.f;                                  // 2^7
#pragma unroll
      for (int f = 7; f < XYZ_F; ++f) {
        float s0, c0, s1, c1, s2, c2;
        __sincosf(cx * fr, &s0, &c0);
        __sincosf(cy * fr, &s1, &c1);
        __sincosf(cz * fr, &s2, &c2);
        F[3 + f * 3 + 0] = (_Float16)s0;  F[33 + f * 3 + 0] = (_Float16)c0;
        F[3 + f * 3 + 1] = (_Float16)s1;  F[33 + f * 3 + 1] = (_Float16)c1;
        F[3 + f * 3 + 2] = (_Float16)s2;  F[33 + f * 3 + 2] = (_Float16)c2;
        fr *= 2.f;
      }
      F[63] = (_Float16)ddx; F[64] = (_Float16)ddy; F[65] = (_Float16)ddz;
      fr = 1.f;
#pragma unroll
      for (int f = 0; f < DIR_F; ++f) {
        float s0, c0, s1, c1, s2, c2;
        __sincosf(ddx * fr, &s0, &c0);
        __sincosf(ddy * fr, &s1, &c1);
        __sincosf(ddz * fr, &s2, &c2);
        F[66 + f * 3 + 0] = (_Float16)s0;  F[78 + f * 3 + 0] = (_Float16)c0;
        F[66 + f * 3 + 1] = (_Float16)s1;  F[78 + f * 3 + 1] = (_Float16)c1;
        F[66 + f * 3 + 2] = (_Float16)s2;  F[78 + f * 3 + 2] = (_Float16)c2;
        fr *= 2.f;
      }
#pragma unroll
      for (int i = IN_DEF; i < KPAD0; ++i) F[i] = (_Float16)0.f;
    }
    // base panorama uv (cheap; computed by all lanes, both halves need it later)
    float rc = sqrtf(cx * cx + cy * cy + cz * cz) + 1e-9f;
    float u_base = atan2f(cx, cz) * 0.31830988618f;                 // /pi
    float sv = fminf(fmaxf(cy / rc, -1.f), 1.f);
    float v_base = asinf(sv) * 0.63661977236f;                      // /(pi/2)
    __syncthreads();

    // ---- deform layer 0: [16x96] x [96x128] ----
    Frag16 a0[3];
#pragma unroll
    for (int kc = 0; kc < 3; ++kc) a0[kc] = load_a(As32, KPAD0 / 2, kc, lane);
    v8f acc[8];
#pragma unroll
    for (int nt = 0; nt < 8; ++nt) acc[nt] = zero8();
#pragma unroll
    for (int nt = 0; nt < 8; ++nt)
#pragma unroll
      for (int kc = 0; kc < 3; ++kc) {
        Frag16 b = load_b(W0t32, KPAD0 / 2, nt * 16, kc, lane);
        acc[nt] = WMMA(a0[kc].h, b.h, acc[nt]);
      }
    __syncthreads();
    store_h_relu(As, acc, db0, lane);
    __syncthreads();

    // ---- deform layer 1: [16x128] x [128x128] ----
    Frag16 a1[4];
#pragma unroll
    for (int kc = 0; kc < 4; ++kc) a1[kc] = load_a(As32, MLPW / 2, kc, lane);
#pragma unroll
    for (int nt = 0; nt < 8; ++nt) acc[nt] = zero8();
#pragma unroll
    for (int nt = 0; nt < 8; ++nt)
#pragma unroll
      for (int kc = 0; kc < 4; ++kc) {
        Frag16 b = load_b(W1t32, MLPW / 2, nt * 16, kc, lane);
        acc[nt] = WMMA(a1[kc].h, b.h, acc[nt]);
      }
    __syncthreads();
    store_h_relu(As, acc, db1, lane);
    __syncthreads();

    // ---- deform layer 2: [16x128] x [128x2 (pad 16)] ----
    Frag16 a2[4];
#pragma unroll
    for (int kc = 0; kc < 4; ++kc) a2[kc] = load_a(As32, MLPW / 2, kc, lane);
    v8f acc2 = zero8();
#pragma unroll
    for (int kc = 0; kc < 4; ++kc) {
      Frag16 b = load_b(W2t32, MLPW / 2, 0, kc, lane);
      acc2 = WMMA(a2[kc].h, b.h, acc2);
    }
    __syncthreads();
    float* duvf = (float*)As;                       // reuse per-wave staging as f32 [16][2]
    {
      int n0 = lane & 15;
      if (n0 < 2) {
#pragma unroll
        for (int j = 0; j < 8; ++j) duvf[(j + 8 * hi) * 2 + n0] = acc2[j] + db2[n0];
      }
    }
    __syncthreads();
    float du = duvf[m16 * 2 + 0];
    float dv = duvf[m16 * 2 + 1];
    __syncthreads();

    // ---- equirect bilinear sample -> k0 f16 [16][32] (pad 12->32) ----
    // Full-wave split: each half-wave samples 6 of the 12 channels.
    {
      float uu = fminf(fmaxf(u_base + du, -1.f), 1.f);
      float vv = fminf(fmaxf(v_base + dv, -1.f), 1.f);
      float fx = fminf(fmaxf((uu + 1.f) * 0.5f * (float)(EQW - 1), 0.f), (float)(EQW - 1));
      float fy = fminf(fmaxf((vv + 1.f) * 0.5f * (float)(EQH - 1), 0.f), (float)(EQH - 1));
      int x0 = (int)fx, y0 = (int)fy;
      int x1 = min(x0 + 1, EQW - 1), y1 = min(y0 + 1, EQH - 1);
      float wx = fx - x0, wy = fy - y0;
      int i00 = y0 * EQW + x0, i01 = y0 * EQW + x1;
      int i10 = y1 * EQW + x0, i11 = y1 * EQW + x1;
      _Float16* K0 = As + m16 * 32;
#pragma unroll
      for (int cc = 0; cc < 6; ++cc) {
        int c = hi * 6 + cc;
        const float* g = equ + c * (EQH * EQW);
        float c0 = g[i00] * (1.f - wx) + g[i01] * wx;
        float c1 = g[i10] * (1.f - wx) + g[i11] * wx;
        K0[c] = (_Float16)(c0 * (1.f - wy) + c1 * wy);
      }
#pragma unroll
      for (int cc = 0; cc < 10; ++cc) K0[RGBD + hi * 10 + cc] = (_Float16)0.f;
    }
    __syncthreads();

    // ---- color layer 0: [16x32] x [32x128] (K padded 12->32) ----
    Frag16 a3 = load_a(As32, 32 / 2, 0, lane);
#pragma unroll
    for (int nt = 0; nt < 8; ++nt) acc[nt] = zero8();
#pragma unroll
    for (int nt = 0; nt < 8; ++nt) {
      Frag16 b = load_b(C0t32, 32 / 2, nt * 16, 0, lane);
      acc[nt] = WMMA(a3.h, b.h, acc[nt]);
    }
    __syncthreads();
    store_h_relu(As, acc, cb0, lane);
    __syncthreads();

    // ---- color layer 1: [16x128] x [128x128] ----
    Frag16 a4[4];
#pragma unroll
    for (int kc = 0; kc < 4; ++kc) a4[kc] = load_a(As32, MLPW / 2, kc, lane);
#pragma unroll
    for (int nt = 0; nt < 8; ++nt) acc[nt] = zero8();
#pragma unroll
    for (int nt = 0; nt < 8; ++nt)
#pragma unroll
      for (int kc = 0; kc < 4; ++kc) {
        Frag16 b = load_b(C1t32, MLPW / 2, nt * 16, kc, lane);
        acc[nt] = WMMA(a4[kc].h, b.h, acc[nt]);
      }
    __syncthreads();
    store_h_relu(As, acc, cb1, lane);
    __syncthreads();

    // ---- color layer 2: [16x128] x [128x3 (pad 16)] + sigmoid -> global ----
    Frag16 a5[4];
#pragma unroll
    for (int kc = 0; kc < 4; ++kc) a5[kc] = load_a(As32, MLPW / 2, kc, lane);
    v8f acc3 = zero8();
#pragma unroll
    for (int kc = 0; kc < 4; ++kc) {
      Frag16 b = load_b(C2t32, MLPW / 2, 0, kc, lane);
      acc3 = WMMA(a5[kc].h, b.h, acc3);
    }
    {
      int n0 = lane & 15;
      if (n0 < 3) {
#pragma unroll
        for (int j = 0; j < 8; ++j) {
          float v = acc3[j] + cb2[n0];
          rgb_out[(p0 + j + 8 * hi) * 3 + n0] = 1.f / (1.f + __expf(-v));
        }
      }
    }
    __syncthreads();
  }
}

// ================= kernel 4: per-ray weighted accumulation =====================
__global__ __launch_bounds__(128) void finalize_kernel(
    const float* __restrict__ wbuf, const float* __restrict__ rgb,
    float* __restrict__ out) {
  int gwave = (blockIdx.x * 128 + threadIdx.x) >> 5;   // one wave per ray
  int lane = threadIdx.x & 31;
  int ray = gwave;
  float s0 = 0.f, s1 = 0.f, s2 = 0.f, aw = 0.f;
  for (int s = lane; s < NSAMP; s += 32) {
    int p = ray * NSAMP + s;
    float w = wbuf[p];
    aw += w;
    s0 += w * rgb[p * 3 + 0];
    s1 += w * rgb[p * 3 + 1];
    s2 += w * rgb[p * 3 + 2];
  }
#pragma unroll
  for (int off = 16; off > 0; off >>= 1) {
    s0 += __shfl_down(s0, off);
    s1 += __shfl_down(s1, off);
    s2 += __shfl_down(s2, off);
    aw += __shfl_down(aw, off);
  }
  if (lane == 0) {
    float bg = 1.f - aw;
    out[ray * 3 + 0] = s0 + bg;
    out[ray * 3 + 1] = s1 + bg;
    out[ray * 3 + 2] = s2 + bg;
  }
}

// =============================== launch ========================================
extern "C" void kernel_launch(void* const* d_in, const int* in_sizes, int n_in,
                              void* d_out, int out_size, void* d_ws, size_t ws_size,
                              hipStream_t stream) {
  (void)in_sizes; (void)n_in; (void)out_size; (void)ws_size;
  const float* rays_o = (const float*)d_in[0];
  const float* rays_d = (const float*)d_in[1];
  const float* dgrid  = (const float*)d_in[2];
  const float* equ    = (const float*)d_in[3];
  const float* dW0 = (const float*)d_in[4];  const float* db0 = (const float*)d_in[5];
  const float* dW1 = (const float*)d_in[6];  const float* db1 = (const float*)d_in[7];
  const float* dW2 = (const float*)d_in[8];  const float* db2 = (const float*)d_in[9];
  const float* cW0 = (const float*)d_in[10]; const float* cb0 = (const float*)d_in[11];
  const float* cW1 = (const float*)d_in[12]; const float* cb1 = (const float*)d_in[13];
  const float* cW2 = (const float*)d_in[14]; const float* cb2 = (const float*)d_in[15];
  float* out = (float*)d_out;

  float* ws = (float*)d_ws;
  float* contracted = ws;                 // 3*NPTS
  float* wbuf       = ws + 3 * NPTS;      // NPTS (alpha, then weights in place)
  float* rgbbuf     = ws + 4 * NPTS;      // 3*NPTS

  sample_density_kernel<<<NPTS / 256, 256, 0, stream>>>(rays_o, rays_d, dgrid, contracted, wbuf);
  composite_weights_kernel<<<(NRAYS + 255) / 256, 256, 0, stream>>>(wbuf);

  (void)hipFuncSetAttribute(reinterpret_cast<const void*>(mlp_fused_kernel),
                            hipFuncAttributeMaxDynamicSharedMemorySize, SMEM_BYTES);
  mlp_fused_kernel<<<NPTS / (16 * 8 * TILES_PER_BLOCK), 256, SMEM_BYTES, stream>>>(
      rays_d, equ, dW0, db0, dW1, db1, dW2, db2, cW0, cb0, cW1, cb1, cW2, cb2,
      contracted, rgbbuf);

  finalize_kernel<<<(NRAYS * 32) / 128, 128, 0, stream>>>(wbuf, rgbbuf, out);
}
